// R2DNet_34411277976336
// MI455X (gfx1250) — matrix operand order repaired and checked by reading, not compiled, not run on hardware
//
#include <hip/hip_runtime.h>
#include <hip/hip_bf16.h>
#include <stdint.h>

typedef __attribute__((ext_vector_type(16))) _Float16 v16h;
typedef __attribute__((ext_vector_type(8)))  float    v8f;

#define NIMG 192   // N*T*M = 2*8*12

union FragU { v16h h; uint32_t u[8]; };
union Vec8H { uint4 v; _Float16 h[8]; };

// ---------------------------------------------------------------------------
// Convert input x: NCHW f32 [192,3,64,64] -> NHWC f16 [192,64,64,3]
// ---------------------------------------------------------------------------
__global__ void k_convert_x(const float* __restrict__ x, _Float16* __restrict__ out) {
    int gid = blockIdx.x * 256 + threadIdx.x;
    const int total = NIMG * 64 * 64 * 3;
    if (gid >= total) return;
    int c = gid % 3;
    int t = gid / 3;
    int px = t % 64; t /= 64;
    int py = t % 64;
    int img = t / 64;
    out[gid] = (_Float16)x[((img * 3 + c) * 64 + py) * 64 + px];
}

// ---------------------------------------------------------------------------
// Reorder conv weights: OIHW f32 [Cout,Cin,3,3] -> f16 [K=9*Cin][Cout]
// k = (ky*3+kx)*Cin + ci  (matches NHWC im2col gather order)
// ---------------------------------------------------------------------------
__global__ void k_convert_wconv(const float* __restrict__ w, _Float16* __restrict__ wk,
                                int Cin, int Cout) {
    int K = 9 * Cin;
    int gid = blockIdx.x * 256 + threadIdx.x;
    if (gid >= K * Cout) return;
    int o = gid % Cout;
    int k = gid / Cout;
    int tap = k / Cin;
    int ci  = k - tap * Cin;
    int ky = tap / 3, kx = tap - ky * 3;
    wk[gid] = (_Float16)w[((o * Cin + ci) * 3 + ky) * 3 + kx];
}

// ---------------------------------------------------------------------------
// Implicit-GEMM 3x3 SAME conv, f16 in/out, f32 accumulate via WMMA.
// M = NIMG*H*W (pixels), N = Cout, K = 9*Cin.
// Workgroup: 256 threads = 8 waves (4 M x 2 N); macro-tile 64(M) x 64(N);
// each wave computes 16x32 (two 16x16 C tiles sharing one A fragment).
// Fast A path (Cin%32==0): one B128 global load + one B128 LDS store/thread.
// ---------------------------------------------------------------------------
__global__ void __launch_bounds__(256) k_conv3x3_wmma(
    const _Float16* __restrict__ in, const _Float16* __restrict__ wk,
    const float* __restrict__ bias, _Float16* __restrict__ out,
    int H, int W, int Cin, int Cout)
{
    __shared__ _Float16 ldsA[64][48];   // rows 96B (16B-aligned for b128)
    __shared__ _Float16 ldsBt[64][48];  // [n][k] transposed for pair loads

    const int K    = 9 * Cin;
    const int HW   = H * W;
    const int Mtot = NIMG * HW;
    const bool fastA = (Cin & 31) == 0;

    const int tid  = threadIdx.x;
    const int lane = tid & 31;
    const int wave = tid >> 5;
    const int wm   = (wave >> 1) * 16;          // 4 wave-rows in M
    const int wn   = (wave & 1) * 32;           // 2 wave-cols in N (32 wide)
    const int l16  = lane & 15;
    const int aHi  = (lane & 16) ? 8  : 0;      // 16-bit A layout: hi half-wave K+8
    const int bHi  = (lane & 16) ? 16 : 0;      // k order used here for A and B: hi half-wave K+16 (contiguous)

    const int m0 = blockIdx.x * 64;
    const int n0 = blockIdx.y * 64;

    v8f acc0 = {};
    v8f acc1 = {};

    for (int k0 = 0; k0 < K; k0 += 32) {
        // prefetch next weight K-slice (global_prefetch_b8)
        if (k0 + 32 < K)
            __builtin_prefetch(&wk[(size_t)(k0 + 32) * Cout + n0 + (tid & 63)], 0, 0);

        // ---- stage A tile 64x32 ----
        if (fastA) {
            // 32-wide K slab of one im2col row = contiguous NHWC run (same tap)
            int mi   = tid >> 2;
            int kseg = (tid & 3) * 8;
            int m = m0 + mi;
            int k = k0 + kseg;
            uint4 val = {0u, 0u, 0u, 0u};
            if (m < Mtot) {
                int img = m / HW;
                int rem = m - img * HW;
                int py = rem / W;
                int px = rem - py * W;
                int tap = k / Cin;
                int ci  = k - tap * Cin;
                int ky = tap / 3, kx = tap - ky * 3;
                int iy = py + ky - 1, ix = px + kx - 1;
                if (iy >= 0 && iy < H && ix >= 0 && ix < W)
                    val = *(const uint4*)&in[((size_t)(img * H + iy) * W + ix) * Cin + ci];
            }
            *(uint4*)&ldsA[mi][kseg] = val;
        } else {
            // generic gather (first layer only: Cin=3, K=27, single iteration)
#pragma unroll
            for (int t = 0; t < 8; ++t) {
                int idx = tid + t * 256;
                int mi = idx >> 5, kk = idx & 31;
                int m = m0 + mi, k = k0 + kk;
                _Float16 v = (_Float16)0.0f;
                if (m < Mtot && k < K) {
                    int img = m / HW;
                    int rem = m - img * HW;
                    int py = rem / W;
                    int px = rem - py * W;
                    int tap = k / Cin;
                    int ci  = k - tap * Cin;
                    int ky = tap / 3, kx = tap - ky * 3;
                    int iy = py + ky - 1, ix = px + kx - 1;
                    if (iy >= 0 && iy < H && ix >= 0 && ix < W)
                        v = in[((size_t)(img * H + iy) * W + ix) * Cin + ci];
                }
                ldsA[mi][kk] = v;
            }
        }
        // ---- stage B tile 32x64 (vector load over N, transpose into LDS) ----
        {
            int kk   = tid >> 3;
            int nseg = (tid & 7) * 8;
            int k = k0 + kk;
            Vec8H bv;
            bv.v = (uint4){0u, 0u, 0u, 0u};
            if (k < K) bv.v = *(const uint4*)&wk[(size_t)k * Cout + n0 + nseg];
#pragma unroll
            for (int j = 0; j < 8; ++j) ldsBt[nseg + j][kk] = bv.h[j];
        }
        __syncthreads();

        // ---- per-lane fragments (contiguous k = 16*half + s for both A and B) ----
        FragU af, bf0, bf1;
        const uint32_t* arow = (const uint32_t*)(&ldsA[wm + l16][0]);
#pragma unroll
        for (int vv = 0; vv < 8; ++vv) {
            int kb = bHi + 2 * vv;
            af.u[vv] = arow[kb >> 1];
        }
        const uint32_t* brow0 = (const uint32_t*)(&ldsBt[wn + l16][0]);
        const uint32_t* brow1 = (const uint32_t*)(&ldsBt[wn + 16 + l16][0]);
#pragma unroll
        for (int vv = 0; vv < 8; ++vv) {
            bf0.u[vv] = brow0[(bHi >> 1) + vv];
            bf1.u[vv] = brow1[(bHi >> 1) + vv];
        }
        acc0 = __builtin_amdgcn_wmma_f32_16x16x32_f16(
                   false, af.h, false, bf0.h, (short)0, acc0, false, false);
        acc1 = __builtin_amdgcn_wmma_f32_16x16x32_f16(
                   false, af.h, false, bf1.h, (short)0, acc1, false, false);
        __syncthreads();
    }

    // ---- epilogue: bias + ReLU, f32 C layout: VGPR r -> M = r (+8 hi half) ----
    const int nA   = n0 + wn + l16;
    const int nB   = nA + 16;
    const int mrow = m0 + wm + ((lane & 16) ? 8 : 0);
    const float bA = bias[nA];
    const float bB = bias[nB];
#pragma unroll
    for (int r = 0; r < 8; ++r) {
        int m = mrow + r;
        if (m < Mtot) {
            float v0 = acc0[r] + bA; v0 = v0 > 0.f ? v0 : 0.f;
            float v1 = acc1[r] + bB; v1 = v1 > 0.f ? v1 : 0.f;
            out[(size_t)m * Cout + nA] = (_Float16)v0;
            out[(size_t)m * Cout + nB] = (_Float16)v1;
        }
    }
}

// ---------------------------------------------------------------------------
// 2x2 maxpool, NHWC f16
// ---------------------------------------------------------------------------
__global__ void k_maxpool(const _Float16* __restrict__ in, _Float16* __restrict__ out,
                          int H, int W, int C)
{
    int H2 = H >> 1, W2 = W >> 1;
    long long total = (long long)NIMG * H2 * W2 * C;
    long long gid = (long long)blockIdx.x * 256 + threadIdx.x;
    if (gid >= total) return;
    int c = (int)(gid % C);
    long long r = gid / C;
    int x = (int)(r % W2); r /= W2;
    int y = (int)(r % H2);
    int img = (int)(r / H2);
    size_t base = (((size_t)img * H + 2 * y) * W + 2 * x) * C + c;
    float a = (float)in[base];
    float b = (float)in[base + C];
    float d = (float)in[base + (size_t)W * C];
    float e = (float)in[base + (size_t)W * C + C];
    out[gid] = (_Float16)fmaxf(fmaxf(a, b), fmaxf(d, e));
}

// ---------------------------------------------------------------------------
// conv1da weights f32 [512,2048] -> f16 [K=2048][N=512]
// ---------------------------------------------------------------------------
__global__ void k_convert_w1(const float* __restrict__ w, _Float16* __restrict__ wk) {
    int gid = blockIdx.x * 256 + threadIdx.x;
    if (gid >= 2048 * 512) return;
    int nn = gid % 512;
    int k  = gid / 512;
    wk[gid] = (_Float16)w[nn * 2048 + k];
}

// ---------------------------------------------------------------------------
// Head GEMM: node1[192,512] = feats[192,2048] @ W1^T via WMMA.
// feats come from final NHWC act [192,2,2,512]; NCHW flatten f = c*4 + pos
// is folded into the A gather: act[m*2048 + (f&3)*512 + (f>>2)].
// Same 64x64 macro-tile / 16x32-per-wave structure as the conv kernel.
// ---------------------------------------------------------------------------
__global__ void __launch_bounds__(256) k_gemm_feats_wmma(
    const _Float16* __restrict__ act, const _Float16* __restrict__ wk,
    float* __restrict__ out)
{
    __shared__ _Float16 ldsA[64][48];
    __shared__ _Float16 ldsBt[64][48];

    const int tid  = threadIdx.x;
    const int lane = tid & 31;
    const int wave = tid >> 5;
    const int wm   = (wave >> 1) * 16;
    const int wn   = (wave & 1) * 32;
    const int l16  = lane & 15;
    const int aHi  = (lane & 16) ? 8  : 0;
    const int bHi  = (lane & 16) ? 16 : 0;
    const int m0 = blockIdx.x * 64;     // 192 = 3 * 64 exact
    const int n0 = blockIdx.y * 64;     // 512 = 8 * 64 exact

    v8f acc0 = {};
    v8f acc1 = {};

    for (int k0 = 0; k0 < 2048; k0 += 32) {
        // A: permuted gather (m always < 192 by grid construction)
#pragma unroll
        for (int t = 0; t < 8; ++t) {
            int idx = tid + t * 256;
            int mi = idx >> 5, kk = idx & 31;
            int m = m0 + mi, k = k0 + kk;
            ldsA[mi][kk] = act[(size_t)m * 2048 + (k & 3) * 512 + (k >> 2)];
        }
        // B: vector load over N, transpose into LDS
        {
            int kk   = tid >> 3;
            int nseg = (tid & 7) * 8;
            Vec8H bv;
            bv.v = *(const uint4*)&wk[(size_t)(k0 + kk) * 512 + n0 + nseg];
#pragma unroll
            for (int j = 0; j < 8; ++j) ldsBt[nseg + j][kk] = bv.h[j];
        }
        __syncthreads();

        FragU af, bf0, bf1;
        const uint32_t* arow = (const uint32_t*)(&ldsA[wm + l16][0]);
#pragma unroll
        for (int vv = 0; vv < 8; ++vv) {
            int kb = bHi + 2 * vv;
            af.u[vv] = arow[kb >> 1];
        }
        const uint32_t* brow0 = (const uint32_t*)(&ldsBt[wn + l16][0]);
        const uint32_t* brow1 = (const uint32_t*)(&ldsBt[wn + 16 + l16][0]);
#pragma unroll
        for (int vv = 0; vv < 8; ++vv) {
            bf0.u[vv] = brow0[(bHi >> 1) + vv];
            bf1.u[vv] = brow1[(bHi >> 1) + vv];
        }
        acc0 = __builtin_amdgcn_wmma_f32_16x16x32_f16(
                   false, af.h, false, bf0.h, (short)0, acc0, false, false);
        acc1 = __builtin_amdgcn_wmma_f32_16x16x32_f16(
                   false, af.h, false, bf1.h, (short)0, acc1, false, false);
        __syncthreads();
    }

    const int nA   = n0 + wn + l16;
    const int nB   = nA + 16;
    const int mrow = m0 + wm + ((lane & 16) ? 8 : 0);
#pragma unroll
    for (int r = 0; r < 8; ++r) {
        int m = mrow + r;
        out[(size_t)m * 512 + nA] = acc0[r];
        out[(size_t)m * 512 + nB] = acc1[r];
    }
}

// ---------------------------------------------------------------------------
// Per-(n,t) ragged head: h = relu(A_hat @ node1); pooled = mask-mean(h @ WL^T).
// One block per nt (16 blocks), 256 threads over the 512 channels.
// ---------------------------------------------------------------------------
__global__ void __launch_bounds__(256) k_head_block(
    const float* __restrict__ node1, const float* __restrict__ dist,
    const int* __restrict__ dist_num, const float* __restrict__ WL,
    float* __restrict__ pooled)
{
    __shared__ float red[256][8];
    int nt  = blockIdx.x;
    int tid = threadIdx.x;
    int n = dist_num[nt];
    n = n < 1 ? 1 : (n > 12 ? 12 : n);
    const float* Ad = dist  + nt * 144;
    const float* nb = node1 + (size_t)nt * 12 * 512;

    float acc[8] = {0.f, 0.f, 0.f, 0.f, 0.f, 0.f, 0.f, 0.f};
    for (int c = tid; c < 512; c += 256) {
        float col[12];
#pragma unroll
        for (int j = 0; j < 12; ++j) col[j] = (j < n) ? nb[j * 512 + c] : 0.f;
        for (int i = 0; i < n; ++i) {
            float h = 0.f;
            for (int j = 0; j < n; ++j) h += Ad[i * 12 + j] * col[j];
            h = fmaxf(h, 0.f);
#pragma unroll
            for (int g = 0; g < 8; ++g) acc[g] += h * WL[g * 512 + c];
        }
    }
#pragma unroll
    for (int g = 0; g < 8; ++g) red[tid][g] = acc[g];
    __syncthreads();
    for (int s = 128; s > 0; s >>= 1) {
        if (tid < s)
            for (int g = 0; g < 8; ++g) red[tid][g] += red[tid + s][g];
        __syncthreads();
    }
    if (tid < 8) pooled[nt * 8 + tid] = red[0][tid] / (float)n;
}

// group_out[n][g] = mean_t pooled_raw_reshape: row n*8+g, col t  (torch quirk)
__global__ void k_final_mean(const float* __restrict__ pooled, float* __restrict__ out) {
    int i = threadIdx.x;
    if (i < 16) {
        float s = 0.f;
        for (int t = 0; t < 8; ++t) s += pooled[i * 8 + t];
        out[i] = s * 0.125f;
    }
}

// ---------------------------------------------------------------------------
extern "C" void kernel_launch(void* const* d_in, const int* in_sizes, int n_in,
                              void* d_out, int out_size, void* d_ws, size_t ws_size,
                              hipStream_t stream)
{
    (void)in_sizes; (void)n_in; (void)out_size; (void)ws_size;

    // d_in order: x, dist, dist_num, 16x(w,b), conv1da_w, convLinear_w
    const float* x        = (const float*)d_in[0];
    const float* dist     = (const float*)d_in[1];
    const int*   dist_num = (const int*)d_in[2];
    const float* w1       = (const float*)d_in[35];
    const float* wl       = (const float*)d_in[36];

    char* ws = (char*)d_ws;
    _Float16* actA = (_Float16*)(ws);                      // 100,663,296 B
    _Float16* actB = (_Float16*)(ws + 100663296u);         // 100,663,296 B
    _Float16* wbuf = (_Float16*)(ws + 201326592u);         //   4,718,592 B
    _Float16* w1k  = (_Float16*)(ws + 206045184u);         //   2,097,152 B
    float*    node1 = (float*)  (ws + 208142336u);         //     393,216 B
    float*    pooled = (float*) (ws + 208535552u);         //         512 B

    _Float16* cur = actA;
    _Float16* nxt = actB;

    k_convert_x<<<(NIMG * 64 * 64 * 3 + 255) / 256, 256, 0, stream>>>(x, cur);

    static const int CFGv[21] = {64,64,0,128,128,0,256,256,256,256,0,
                                 512,512,512,512,0,512,512,512,512,0};
    int Cin = 3, H = 64, W = 64, li = 0;
    for (int s = 0; s < 21; ++s) {
        int v = CFGv[s];
        if (v == 0) {
            int H2 = H / 2, W2 = W / 2;
            long long tot = (long long)NIMG * H2 * W2 * Cin;
            k_maxpool<<<(int)((tot + 255) / 256), 256, 0, stream>>>(cur, nxt, H, W, Cin);
            H = H2; W = W2;
            _Float16* t = cur; cur = nxt; nxt = t;
        } else {
            int Cout = v, K = 9 * Cin;
            const float* wcv = (const float*)d_in[3 + 2 * li];
            const float* bcv = (const float*)d_in[4 + 2 * li];
            k_convert_wconv<<<(K * Cout + 255) / 256, 256, 0, stream>>>(wcv, wbuf, Cin, Cout);
            dim3 g((NIMG * H * W + 63) / 64, Cout / 64);
            k_conv3x3_wmma<<<g, 256, 0, stream>>>(cur, wbuf, bcv, nxt, H, W, Cin, Cout);
            Cin = Cout; ++li;
            _Float16* t = cur; cur = nxt; nxt = t;
        }
    }

    // Head: 192x512x2048 GEMM (WMMA) + ragged GCN + pools
    k_convert_w1<<<4096, 256, 0, stream>>>(w1, w1k);
    dim3 g2(3, 8);
    k_gemm_feats_wmma<<<g2, 256, 0, stream>>>(cur, w1k, node1);
    k_head_block<<<16, 256, 0, stream>>>(node1, dist, dist_num, wl, pooled);
    k_final_mean<<<1, 32, 0, stream>>>(pooled, (float*)d_out);
}
